// AttentionPooling_42571715837971
// MI455X (gfx1250) — matrix operand order repaired and checked by reading, not compiled, not run on hardware
//
#include <hip/hip_runtime.h>
#include <math.h>

// ---------------------------------------------------------------------------
// AttentionPooling on MI455X (gfx1250, wave32, WMMA)
//   s = tanh(x@W1 + b1) @ w2 + b2 ; segment softmax ; pooled = seg_sum(alpha*x)
//   out = pooled @ Wp + bp
// Heavy matmul (1e6x256 @ 256x256) runs on v_wmma_f32_16x16x32_bf16 with W1
// staged in LDS in fragment layout; all 8 B fragments of an N-tile are loaded
// as one batch (sched_barrier) so the 8-deep WMMA chain runs behind one wait.
// ---------------------------------------------------------------------------

typedef __attribute__((ext_vector_type(16))) __bf16 v16bf;
typedef __attribute__((ext_vector_type(8)))  float  v8f;

#define DDIM   256
#define NTILES 16            // DDIM / 16
#define KTILES 8             // DDIM / 32
#define FRAG_DW (NTILES*KTILES*32*8)   // 32768 dwords = 128 KB per matrix

__device__ __forceinline__ void sched_fence() {
#if __has_builtin(__builtin_amdgcn_sched_barrier)
  __builtin_amdgcn_sched_barrier(0);   // nothing crosses: keep load batch ahead
#endif
}

__device__ __forceinline__ unsigned pack_bf16(float a, float b) {
#if __has_builtin(__builtin_amdgcn_cvt_pk_bf16_f32)
  typedef __attribute__((ext_vector_type(2))) __bf16 v2bf;
  union { v2bf v; unsigned u; } c;
  c.v = __builtin_amdgcn_cvt_pk_bf16_f32(a, b);
  return c.u;
#else
  union { __bf16 h; unsigned short u; } ca, cb;
  ca.h = (__bf16)a; cb.h = (__bf16)b;
  return (unsigned)ca.u | ((unsigned)cb.u << 16);
#endif
}

__device__ __forceinline__ float fast_tanh(float v) {
#if __has_builtin(__builtin_amdgcn_tanhf)
  return __builtin_amdgcn_tanhf(v);     // v_tanh_f32 (gfx1250 trans op)
#else
  return tanhf(v);
#endif
}

// Build one 16x32 bf16 A fragment from 16 fp32 values per lane: p[0..7] and p[16..23].
// A layout: lane 0-15 = row M (K 0-7 / 16-23), lane 16-31 = row M (K 8-15 / 24-31).
__device__ __forceinline__ v16bf make_a_frag(const float* p) {
  float4 f0 = *(const float4*)(p);
  float4 f1 = *(const float4*)(p + 4);
  float4 g0 = *(const float4*)(p + 16);
  float4 g1 = *(const float4*)(p + 20);
  union { unsigned u[8]; v16bf v; } a;
  a.u[0] = pack_bf16(f0.x, f0.y); a.u[1] = pack_bf16(f0.z, f0.w);
  a.u[2] = pack_bf16(f1.x, f1.y); a.u[3] = pack_bf16(f1.z, f1.w);
  a.u[4] = pack_bf16(g0.x, g0.y); a.u[5] = pack_bf16(g0.z, g0.w);
  a.u[6] = pack_bf16(g1.x, g1.y); a.u[7] = pack_bf16(g1.z, g1.w);
  return a.v;
}

// Pack W (row-major KxN = 256x256 fp32) into WMMA B-fragment order:
// dword index = ((nt*8 + kt)*32 + lane)*8 + i
//   lane: n = nt*16 + (lane&15), koff = (lane<16) ? 0 : 8
//   dword i holds bf16 pair (K = kt*32+koff+2i+(i>=4?8:0), +1) for column n.
extern "C" __global__ __launch_bounds__(256)
void ap_prep_frags(const float* __restrict__ W1, const float* __restrict__ Wp,
                   unsigned* __restrict__ w1f, unsigned* __restrict__ wpf) {
  int tid = blockIdx.x * blockDim.x + threadIdx.x;   // 0 .. 2*FRAG_DW-1
  int which = tid >> 15;
  int t = tid & (FRAG_DW - 1);
  int i    = t & 7;
  int lane = (t >> 3) & 31;
  int kt   = (t >> 8) & 7;
  int nt   = (t >> 11) & 15;
  const float* W = which ? Wp : W1;
  unsigned* dst  = which ? wpf : w1f;
  int n    = nt * 16 + (lane & 15);
  int koff = (lane < 16) ? 0 : 8;
  int k    = kt * 32 + koff + 2 * i + (i >= 4 ? 8 : 0);
  dst[t] = pack_bf16(W[(size_t)k * DDIM + n], W[(size_t)(k + 1) * DDIM + n]);
}

// Score kernel: 8 waves/block, 16 rows/wave. W1 fragments staged in LDS.
extern "C" __global__ __launch_bounds__(256)
void ap_score(const float* __restrict__ x, const unsigned* __restrict__ w1f,
              const float* __restrict__ b1, const float* __restrict__ w2,
              const float* __restrict__ b2p, float* __restrict__ s_out, int N) {
  extern __shared__ unsigned smem[];           // 128 KB of W1 fragments
  const int tid = threadIdx.x;
  {
    const uint4* src = (const uint4*)w1f;
    uint4* dst = (uint4*)smem;
#pragma unroll
    for (int j = 0; j < FRAG_DW / 4 / 256; ++j)
      dst[tid + j * 256] = src[tid + j * 256];
  }
  __syncthreads();

  const int wave = tid >> 5;
  const int lane = tid & 31;
  const int r0   = blockIdx.x * 128 + wave * 16;
  const int mrow = lane & 15;
  int row = r0 + mrow; if (row > N - 1) row = N - 1;   // clamp, keep EXEC full
  const int koffA = (lane < 16) ? 0 : 8;               // A-layout K split
  const float* xr = x + (size_t)row * DDIM;

  // 8 A fragments (16x32 bf16 each) covering this wave's 16 rows.
  v16bf A[KTILES];
#pragma unroll
  for (int kt = 0; kt < KTILES; ++kt)
    A[kt] = make_a_frag(xr + kt * 32 + koffA);

  float sacc[8];
#pragma unroll
  for (int i = 0; i < 8; ++i) sacc[i] = 0.f;

  const v16bf* bfr = (const v16bf*)smem;
#pragma unroll 1
  for (int nt = 0; nt < NTILES; ++nt) {
    // batch-issue all 16 ds_load_b128 for this N-tile, then run the WMMA chain
    v16bf Bf[KTILES];
#pragma unroll
    for (int kt = 0; kt < KTILES; ++kt)
      Bf[kt] = bfr[(nt * KTILES + kt) * 32 + lane];
    sched_fence();
    v8f c = {};
#pragma unroll
    for (int kt = 0; kt < KTILES; ++kt)
      c = __builtin_amdgcn_wmma_f32_16x16x32_bf16(false, A[kt], false, Bf[kt],
                                                  (short)0, c, false, false);
    // fuse: h = tanh(c + b1[n]); partial s += h * w2[n]  (n = this lane's col)
    int n = nt * 16 + mrow;
    float w2n = w2[n], b1n = b1[n];
#pragma unroll
    for (int i = 0; i < 8; ++i) sacc[i] += fast_tanh(c[i] + b1n) * w2n;
  }

  // reduce over the 16 columns held by this half-wave (C layout: lanes 0-15
  // hold rows 0-7, lanes 16-31 hold rows 8-15)
#pragma unroll
  for (int off = 1; off < 16; off <<= 1)
#pragma unroll
    for (int i = 0; i < 8; ++i) sacc[i] += __shfl_xor(sacc[i], off, 32);

  if (mrow == 0) {
    const float b2v = b2p[0];
    const int mbase = (lane < 16) ? 0 : 8;
#pragma unroll
    for (int i = 0; i < 8; ++i) {
      int r = r0 + mbase + i;
      if (r < N) s_out[r] = sacc[i] + b2v;
    }
  }
}

// Segment starts via binary search in the sorted batch array.
extern "C" __global__ __launch_bounds__(256)
void ap_seg_bounds(const int* __restrict__ batch, int* __restrict__ seg,
                   int N, int B) {
  int b = blockIdx.x * blockDim.x + threadIdx.x;
  if (b > B) return;
  int lo = 0, hi = N;
  while (lo < hi) { int mid = (lo + hi) >> 1; if (batch[mid] < b) lo = mid + 1; else hi = mid; }
  seg[b] = lo;
}

// Per-segment softmax + weighted pooling. One block per segment, thread = dim.
extern "C" __global__ __launch_bounds__(256)
void ap_segpool(const float* __restrict__ s, const int* __restrict__ seg,
                const float* __restrict__ x, float* __restrict__ pooled) {
  __shared__ float red[256];
  const int b = blockIdx.x, tid = threadIdx.x;
  const int beg = seg[b], end = seg[b + 1];

  float m = -INFINITY;
  for (int i = beg + tid; i < end; i += 256) m = fmaxf(m, s[i]);
  red[tid] = m; __syncthreads();
  for (int st = 128; st > 0; st >>= 1) {
    if (tid < st) red[tid] = fmaxf(red[tid], red[tid + st]);
    __syncthreads();
  }
  const float mx = red[0]; __syncthreads();

  float sum = 0.f;
  for (int i = beg + tid; i < end; i += 256) sum += __expf(s[i] - mx);
  red[tid] = sum; __syncthreads();
  for (int st = 128; st > 0; st >>= 1) {
    if (tid < st) red[tid] += red[tid + st];
    __syncthreads();
  }
  const float inv = (end > beg) ? 1.0f / red[0] : 0.f;

  float acc = 0.f;
  for (int i = beg; i < end; ++i) {
    float a = __expf(s[i] - mx) * inv;          // uniform per iteration
    acc += a * x[(size_t)i * DDIM + tid];       // coalesced across threads
  }
  pooled[(size_t)b * DDIM + tid] = acc;
}

// out = pooled @ Wp + bp  (4096x256 @ 256x256), WMMA bf16, B frags from L2.
extern "C" __global__ __launch_bounds__(256)
void ap_out_gemm(const float* __restrict__ pooled, const unsigned* __restrict__ wpf,
                 const float* __restrict__ bp, float* __restrict__ out) {
  const int tid = threadIdx.x;
  const int wave = tid >> 5, lane = tid & 31;
  const int r0   = blockIdx.x * 128 + wave * 16;
  const int mrow = lane & 15;
  const int koffA = (lane < 16) ? 0 : 8;
  const float* pr = pooled + (size_t)(r0 + mrow) * DDIM;

  v16bf A[KTILES];
#pragma unroll
  for (int kt = 0; kt < KTILES; ++kt)
    A[kt] = make_a_frag(pr + kt * 32 + koffA);

  const v16bf* bfr = (const v16bf*)wpf;
  const int mbase = (lane < 16) ? 0 : 8;
#pragma unroll 1
  for (int nt = 0; nt < NTILES; ++nt) {
    v16bf Bf[KTILES];
#pragma unroll
    for (int kt = 0; kt < KTILES; ++kt)
      Bf[kt] = bfr[(nt * KTILES + kt) * 32 + lane];
    sched_fence();
    v8f c = {};
#pragma unroll
    for (int kt = 0; kt < KTILES; ++kt)
      c = __builtin_amdgcn_wmma_f32_16x16x32_bf16(false, A[kt], false, Bf[kt],
                                                  (short)0, c, false, false);
    int n = nt * 16 + mrow;
    float bpn = bp[n];
#pragma unroll
    for (int i = 0; i < 8; ++i)
      out[(size_t)(r0 + mbase + i) * DDIM + n] = c[i] + bpn;
  }
}

extern "C" void kernel_launch(void* const* d_in, const int* in_sizes, int n_in,
                              void* d_out, int out_size, void* d_ws, size_t ws_size,
                              hipStream_t stream) {
  const float* x     = (const float*)d_in[0];
  const int*   batch = (const int*)d_in[1];
  const float* W1    = (const float*)d_in[2];
  const float* b1    = (const float*)d_in[3];
  const float* w2    = (const float*)d_in[4];
  const float* b2    = (const float*)d_in[5];
  const float* Wp    = (const float*)d_in[6];
  const float* bp    = (const float*)d_in[7];
  float* out = (float*)d_out;

  const int N = in_sizes[0] / DDIM;      // 1,000,000
  const int B = out_size / DDIM;         // 4096

  // workspace carve-up (~8.5 MB): W1 frags | Wp frags | s | seg starts | pooled
  char* w = (char*)d_ws;
  unsigned* w1f = (unsigned*)w;                              // 128 KB
  unsigned* wpf = (unsigned*)(w + 131072);                   // 128 KB
  float* s      = (float*)(w + 262144);                      // N floats
  size_t off = 262144 + (((size_t)N * 4 + 255) & ~(size_t)255);
  int* seg      = (int*)(w + off);                           // B+1 ints
  off += (((size_t)(B + 1) * 4 + 255) & ~(size_t)255);
  float* pooled = (float*)(w + off);                         // B*D floats

  ap_prep_frags<<<(2 * FRAG_DW) / 256, 256, 0, stream>>>(W1, Wp, w1f, wpf);

  const int score_blocks = (N + 127) / 128;
  ap_score<<<score_blocks, 256, FRAG_DW * 4, stream>>>(x, w1f, b1, w2, b2, s, N);

  ap_seg_bounds<<<(B + 1 + 255) / 256, 256, 0, stream>>>(batch, seg, N, B);

  ap_segpool<<<B, 256, 0, stream>>>(s, seg, x, pooled);

  ap_out_gemm<<<B / 128, 256, 0, stream>>>(pooled, wpf, bp, out);
}
